// Net_91096256348494
// MI455X (gfx1250) — compile-verified, matrix operand order checked by reading, not run on hardware
//
#include <hip/hip_runtime.h>
#include <hip/hip_bf16.h>
#include <math.h>

// ---------------------------------------------------------------------------
// MI455X (gfx1250) implementation of the DGCNN-ish reference.
// All dense linear layers run through v_wmma_f32_16x16x32_f16 (f16 in, f32 acc).
// wave32 assumed throughout (gfx1250 is wave32-only).
// ---------------------------------------------------------------------------

typedef __attribute__((ext_vector_type(16))) _Float16 v16h;
typedef __attribute__((ext_vector_type(8)))  _Float16 v8h_t;
typedef __attribute__((ext_vector_type(8)))  float    v8f;

#define DEVINL __device__ __forceinline__

static constexpr int BB   = 8;      // batches
static constexpr int NN   = 2048;   // points per batch
static constexpr int NP   = BB * NN; // 16384 total points
static constexpr int KNB  = 16;     // k for kNN
static constexpr float BNSCALE = 0.9999950000374996f; // 1/sqrt(1+1e-5)

// ------------------------- WMMA fragment helpers ---------------------------
// 16-bit A-matrix 16x32 layout (ISA 7.12.2): lanes 0-15 hold row M=lane with
// K = {0..7, 16..23}; lanes 16-31 hold row M=lane-16 with K = {8..15, 24..31}.
// B operand uses the same per-lane pattern with "row" = output column (i.e.
// row of W stored [N,K] row-major). Both loads are two 16B contiguous chunks.

DEVINL v16h load_frag(const _Float16* rowp, int k0, int half) {
  v8h_t lo = *(const v8h_t*)(rowp + k0 + half * 8);
  v8h_t hi = *(const v8h_t*)(rowp + k0 + 16 + half * 8);
  v16h r;
#pragma unroll
  for (int e = 0; e < 8; ++e) { r[e] = lo[e]; r[8 + e] = hi[e]; }
  return r;
}

// ------------------------------- GEMM kernel -------------------------------
// C[M,N] = act( A[M,Kp] * W[N,Kp]^T + bias )
//   mode 0: y = x + bias
//   mode 1: y = bn(relu(x+bias))   (torch MLP block: Linear->ReLU->BN)
//   mode 2: y = relu(bn(x+bias))   (STN conv/fc blocks)
// Block = 256 threads = 8 waves; each wave owns a 16(M) x 64(N) tile:
// 4 f32 accumulators, A fragment reused 4x, K loop software-pipelined.
// Out-of-range rows are CLAMPED on load (stores are masked in the epilogue),
// so all fragment loads are unconditional b128 pairs -> no exec juggling.
__global__ void k_gemm(const _Float16* __restrict__ A,
                       const _Float16* __restrict__ W,
                       const float* __restrict__ bias,
                       const float* __restrict__ bng,
                       const float* __restrict__ bnb,
                       float* __restrict__ outF,
                       _Float16* __restrict__ outH,
                       int M, int N, int Kp, int mode) {
  const int lane = threadIdx.x & 31;
  const int wave = threadIdx.x >> 5;
  const int lr   = lane & 15;
  const int half = lane >> 4;
  const int m0   = blockIdx.y * 128 + wave * 16;
  const int n0   = blockIdx.x * 64;

  const int arow = min(m0 + lr, M - 1);
  const _Float16* arp = A + (size_t)arow * Kp;
  const _Float16* wrp0 = W + (size_t)min(n0 +  0 + lr, N - 1) * Kp;
  const _Float16* wrp1 = W + (size_t)min(n0 + 16 + lr, N - 1) * Kp;
  const _Float16* wrp2 = W + (size_t)min(n0 + 32 + lr, N - 1) * Kp;
  const _Float16* wrp3 = W + (size_t)min(n0 + 48 + lr, N - 1) * Kp;

  v8f c0, c1, c2, c3;
#pragma unroll
  for (int i = 0; i < 8; ++i) { c0[i] = 0.f; c1[i] = 0.f; c2[i] = 0.f; c3[i] = 0.f; }

  // prologue: first K tile
  v16h a  = load_frag(arp, 0, half);
  v16h w0 = load_frag(wrp0, 0, half);
  v16h w1 = load_frag(wrp1, 0, half);
  v16h w2 = load_frag(wrp2, 0, half);
  v16h w3 = load_frag(wrp3, 0, half);

  for (int k0 = 32; k0 < Kp; k0 += 32) {
    __builtin_prefetch(arp + k0 + 32, 0, 3);   // near-cache prefetch of next A tile
    v16h an  = load_frag(arp, k0, half);
    v16h wn0 = load_frag(wrp0, k0, half);
    v16h wn1 = load_frag(wrp1, k0, half);
    v16h wn2 = load_frag(wrp2, k0, half);
    v16h wn3 = load_frag(wrp3, k0, half);
    c0 = __builtin_amdgcn_wmma_f32_16x16x32_f16(false, a, false, w0, (short)0, c0, false, false);
    c1 = __builtin_amdgcn_wmma_f32_16x16x32_f16(false, a, false, w1, (short)0, c1, false, false);
    c2 = __builtin_amdgcn_wmma_f32_16x16x32_f16(false, a, false, w2, (short)0, c2, false, false);
    c3 = __builtin_amdgcn_wmma_f32_16x16x32_f16(false, a, false, w3, (short)0, c3, false, false);
    a = an; w0 = wn0; w1 = wn1; w2 = wn2; w3 = wn3;
  }
  c0 = __builtin_amdgcn_wmma_f32_16x16x32_f16(false, a, false, w0, (short)0, c0, false, false);
  c1 = __builtin_amdgcn_wmma_f32_16x16x32_f16(false, a, false, w1, (short)0, c1, false, false);
  c2 = __builtin_amdgcn_wmma_f32_16x16x32_f16(false, a, false, w2, (short)0, c2, false, false);
  c3 = __builtin_amdgcn_wmma_f32_16x16x32_f16(false, a, false, w3, (short)0, c3, false, false);

  // epilogue: C/D layout -> lane<16: row m0+v, lane>=16: row m0+8+v; col n0+16t+lr
  const float* cs[4] = {nullptr, nullptr, nullptr, nullptr}; (void)cs;
#pragma unroll
  for (int t = 0; t < 4; ++t) {
    const int col = n0 + t * 16 + lr;
    if (col >= N) continue;
    const float bs = bias ? bias[col] : 0.0f;
    float sc = 1.0f, sh = 0.0f;
    if (mode != 0) { sc = bng[col] * BNSCALE; sh = bnb[col]; }
    const v8f* cp = (t == 0) ? &c0 : (t == 1) ? &c1 : (t == 2) ? &c2 : &c3;
#pragma unroll
    for (int v = 0; v < 8; ++v) {
      const int row = m0 + half * 8 + v;
      if (row >= M) continue;
      float x = (*cp)[v] + bs;
      float y;
      if      (mode == 1) y = sc * fmaxf(x, 0.0f) + sh;
      else if (mode == 2) y = fmaxf(sc * x + sh, 0.0f);
      else                y = x;
      if (outF) outF[(size_t)row * N + col] = y;
      if (outH) outH[(size_t)row * N + col] = (_Float16)y;
    }
  }
}

// ------------------------- small utility kernels ---------------------------

// convert f32 [rows,cols] -> f16 [rows,colsp] with zero padding
__global__ void k_cvt(const float* __restrict__ src, _Float16* __restrict__ dst,
                      int rows, int cols, int colsp) {
  int t = blockIdx.x * blockDim.x + threadIdx.x;
  if (t >= rows * colsp) return;
  int r = t / colsp, c = t - r * colsp;
  dst[t] = (c < cols) ? (_Float16)src[(size_t)r * cols + c] : (_Float16)0.0f;
}

// kNN feature build: f = [X[n,:C], 0.01*seq[n]] zero-padded to Dp; sq = |f|^2
__global__ void k_feat(const float* __restrict__ X, int C,
                       const int* __restrict__ seq,
                       _Float16* __restrict__ F, int Dp,
                       float* __restrict__ sq) {
  int n = blockIdx.x * blockDim.x + threadIdx.x;
  if (n >= NP) return;
  _Float16* o = F + (size_t)n * Dp;
  float s = 0.0f;
  for (int c = 0; c < C; ++c) { float v = X[(size_t)n * C + c]; o[c] = (_Float16)v; s += v * v; }
  float tf = 0.01f * (float)seq[n];
  o[C] = (_Float16)tf; s += tf * tf;
  for (int d = C + 1; d < Dp; ++d) o[d] = (_Float16)0.0f;
  sq[n] = s;
}

union H2U { unsigned u; _Float16 h[2]; };

// per-point brute-force kNN (k=16, within batch, excluding self)
template <int DP>
__global__ void k_knn(const _Float16* __restrict__ F,
                      const float* __restrict__ sq,
                      int* __restrict__ idx) {
  int n = blockIdx.x * blockDim.x + threadIdx.x;
  if (n >= NP) return;
  const int b0 = (n >> 11) << 11;
  unsigned fr[DP / 2];
  const unsigned* fp = (const unsigned*)(F + (size_t)n * DP);
#pragma unroll
  for (int d = 0; d < DP / 2; ++d) fr[d] = fp[d];
  const float sqi = sq[n];

  float bd[16]; int bi[16];
#pragma unroll
  for (int s = 0; s < 16; ++s) { bd[s] = 3.4e38f; bi[s] = n; }

  for (int j = 0; j < NN; ++j) {
    int jj = b0 + j;
    if (jj == n) continue;
    const unsigned* gp = (const unsigned*)(F + (size_t)jj * DP);
    float dot = 0.0f;
#pragma unroll
    for (int d = 0; d < DP / 2; ++d) {
      H2U a; a.u = fr[d];
      H2U c; c.u = gp[d];
      dot += (float)a.h[0] * (float)c.h[0] + (float)a.h[1] * (float)c.h[1];
    }
    float dist = sqi + sq[jj] - 2.0f * dot;
    float wmax = bd[0]; int wi = 0;
#pragma unroll
    for (int s = 1; s < 16; ++s) { if (bd[s] > wmax) { wmax = bd[s]; wi = s; } }
    if (dist < wmax) {
#pragma unroll
      for (int s = 0; s < 16; ++s) { if (s == wi) { bd[s] = dist; bi[s] = jj; } }
    }
  }
#pragma unroll
  for (int s = 0; s < 16; ++s) idx[(size_t)n * KNB + s] = bi[s];
}

// build edge features for one chunk of 2048 points:
// Eh[eid, 0:C]=xi, [C:2C]=xj-xi, [2C:Kp]=0   (eid = local_point*16 + k)
__global__ void k_gather(const float* __restrict__ X, int C,
                         const int* __restrict__ idx, int pt0,
                         _Float16* __restrict__ Eh, int Kp) {
  int eid = blockIdx.x * blockDim.x + threadIdx.x;
  if (eid >= NN * KNB) return;
  int p = eid >> 4, k = eid & 15;
  int n = pt0 + p;
  int j = idx[(size_t)n * KNB + k];
  const float* xi = X + (size_t)n * C;
  const float* xj = X + (size_t)j * C;
  _Float16* o = Eh + (size_t)eid * Kp;
  for (int c = 0; c < C; ++c) {
    float a = xi[c];
    o[c]     = (_Float16)a;
    o[C + c] = (_Float16)(xj[c] - a);
  }
  for (int c = 2 * C; c < Kp; ++c) o[c] = (_Float16)0.0f;
}

// attention combine: w = softmax(e @ Att, over k); x_out = sum_k w[k,h(f)]*e[k,f]
// one wave per point (8 points per 256-thread block), one chunk of 2048 points.
template <int F>
__global__ void k_att(const float* __restrict__ e,
                      const float* __restrict__ Att,
                      float* __restrict__ Xout, int pt0) {
  __shared__ float lg[8][16][8];
  __shared__ float wx[8][16][8];
  const int w = threadIdx.x >> 5;
  const int lane = threadIdx.x & 31;
  const int p = blockIdx.x * 8 + w;
  const float* ep = e + (size_t)p * KNB * F;

#pragma unroll
  for (int t = 0; t < 4; ++t) {
    int id = lane + 32 * t;            // 128 (k,h) logits per point
    int k = id >> 3, h = id & 7;
    float acc = 0.0f;
#pragma unroll 8
    for (int f = 0; f < F; ++f) acc += ep[k * F + f] * Att[f * 8 + h];
    lg[w][k][h] = acc;
  }
  __syncthreads();
  if (lane < 8) {
    int h = lane;
    float mx = -3.4e38f;
#pragma unroll
    for (int k = 0; k < KNB; ++k) mx = fmaxf(mx, lg[w][k][h]);
    float sum = 0.0f;
#pragma unroll
    for (int k = 0; k < KNB; ++k) { float ex = expf(lg[w][k][h] - mx); wx[w][k][h] = ex; sum += ex; }
    float inv = 1.0f / sum;
#pragma unroll
    for (int k = 0; k < KNB; ++k) wx[w][k][h] *= inv;
  }
  __syncthreads();
  const int FH = F / 8;
  float* op = Xout + (size_t)(pt0 + p) * F;
  for (int f = lane; f < F; f += 32) {
    int h = f / FH;
    float acc = 0.0f;
#pragma unroll
    for (int k = 0; k < KNB; ++k) acc += wx[w][k][h] * ep[k * F + f];
    op[f] = acc;
  }
}

// global max pool over N points per batch: g[b,c] = max_n X[b*N+n, c]
__global__ void k_maxpool(const float* __restrict__ X, float* __restrict__ g, int C) {
  int t = blockIdx.x * blockDim.x + threadIdx.x;
  if (t >= BB * C) return;
  int b = t / C, c = t - b * C;
  const float* xp = X + (size_t)b * NN * C + c;
  float m = xp[0];
  for (int n = 1; n < NN; ++n) m = fmaxf(m, xp[(size_t)n * C]);
  g[t] = m;
}

__global__ void k_addeye(float* __restrict__ t9) {
  int t = blockIdx.x * blockDim.x + threadIdx.x;
  if (t >= BB * 9) return;
  int i = t % 9;
  if (i == 0 || i == 4 || i == 8) t9[t] += 1.0f;
}

// posT[n,:] = pos[n,:] @ trans[b]  (trans row-major 3x3)
__global__ void k_bmm(const float* __restrict__ pos, const float* __restrict__ t9,
                      float* __restrict__ posT) {
  int n = blockIdx.x * blockDim.x + threadIdx.x;
  if (n >= NP) return;
  int b = n >> 11;
  const float* T = t9 + b * 9;
  float p0 = pos[n * 3 + 0], p1 = pos[n * 3 + 1], p2 = pos[n * 3 + 2];
#pragma unroll
  for (int e = 0; e < 3; ++e)
    posT[n * 3 + e] = p0 * T[0 * 3 + e] + p1 * T[1 * 3 + e] + p2 * T[2 * 3 + e];
}

// concat [x1|x2|x3] -> f16 [NP,448]
__global__ void k_concat(const float* __restrict__ x1, const float* __restrict__ x2,
                         const float* __restrict__ x3, _Float16* __restrict__ o) {
  int t = blockIdx.x * blockDim.x + threadIdx.x;
  if (t >= NP * 448) return;
  int n = t / 448, c = t - n * 448;
  float v;
  if (c < 64)       v = x1[(size_t)n * 64 + c];
  else if (c < 192) v = x2[(size_t)n * 128 + (c - 64)];
  else              v = x3[(size_t)n * 256 + (c - 192)];
  o[t] = (_Float16)v;
}

__global__ void k_logsoftmax(const float* __restrict__ in, float* __restrict__ out) {
  int b = threadIdx.x;
  if (b >= BB) return;
  const float* r = in + b * 60;
  float mx = r[0];
  for (int i = 1; i < 60; ++i) mx = fmaxf(mx, r[i]);
  float s = 0.0f;
  for (int i = 0; i < 60; ++i) s += expf(r[i] - mx);
  float l = logf(s);
  for (int i = 0; i < 60; ++i) out[b * 60 + i] = r[i] - mx - l;
}

// ------------------------------- host side ---------------------------------

static inline void launch_gemm(hipStream_t s, const _Float16* A, const _Float16* W,
                               const float* bias, const float* g, const float* b2,
                               float* oF, _Float16* oH, int M, int N, int Kp, int mode) {
  dim3 grid((N + 63) / 64, (M + 127) / 128);
  k_gemm<<<grid, 256, 0, s>>>(A, W, bias, g, b2, oF, oH, M, N, Kp, mode);
}

static inline void launch_cvt(hipStream_t s, const float* src, _Float16* dst,
                              int rows, int cols, int colsp) {
  int tot = rows * colsp;
  k_cvt<<<(tot + 255) / 256, 256, 0, s>>>(src, dst, rows, cols, colsp);
}

extern "C" void kernel_launch(void* const* d_in, const int* in_sizes, int n_in,
                              void* d_out, int out_size, void* d_ws, size_t ws_size,
                              hipStream_t stream) {
  (void)in_sizes; (void)n_in; (void)out_size; (void)ws_size;

  // -------- input map (depth-first flatten of setup_inputs dict) ----------
  const float* pos   = (const float*)d_in[0];   // [16384,3]
  const int*   seq   = (const int*)d_in[1];     // [16384]
  // stn
  const float* Wc1 = (const float*)d_in[2];  const float* bc1 = (const float*)d_in[3];
  const float* g1  = (const float*)d_in[4];  const float* bb1 = (const float*)d_in[5];
  const float* Wc2 = (const float*)d_in[6];  const float* bc2 = (const float*)d_in[7];
  const float* g2  = (const float*)d_in[8];  const float* bb2 = (const float*)d_in[9];
  const float* Wc3 = (const float*)d_in[10]; const float* bc3 = (const float*)d_in[11];
  const float* g3  = (const float*)d_in[12]; const float* bb3 = (const float*)d_in[13];
  const float* Wf1 = (const float*)d_in[14]; const float* bf1 = (const float*)d_in[15];
  const float* g4  = (const float*)d_in[16]; const float* bb4 = (const float*)d_in[17];
  const float* Wf2 = (const float*)d_in[18]; const float* bf2 = (const float*)d_in[19];
  const float* g5  = (const float*)d_in[20]; const float* bb5 = (const float*)d_in[21];
  const float* Wf3 = (const float*)d_in[22]; const float* bf3 = (const float*)d_in[23];
  // mlp1 (3 blocks)
  const float* Wm10 = (const float*)d_in[24]; const float* bm10 = (const float*)d_in[25];
  const float* gm10 = (const float*)d_in[26]; const float* cm10 = (const float*)d_in[27];
  const float* Wm11 = (const float*)d_in[28]; const float* bm11 = (const float*)d_in[29];
  const float* gm11 = (const float*)d_in[30]; const float* cm11 = (const float*)d_in[31];
  const float* Wm12 = (const float*)d_in[32]; const float* bm12 = (const float*)d_in[33];
  const float* gm12 = (const float*)d_in[34]; const float* cm12 = (const float*)d_in[35];
  const float* att1 = (const float*)d_in[36];
  const float* Wm2  = (const float*)d_in[37]; const float* bm2 = (const float*)d_in[38];
  const float* gm2  = (const float*)d_in[39]; const float* cm2 = (const float*)d_in[40];
  const float* att2 = (const float*)d_in[41];
  const float* Wm3  = (const float*)d_in[42]; const float* bm3 = (const float*)d_in[43];
  const float* gm3  = (const float*)d_in[44]; const float* cm3 = (const float*)d_in[45];
  const float* att3 = (const float*)d_in[46];
  const float* Wl1  = (const float*)d_in[47]; const float* bl1 = (const float*)d_in[48];
  const float* gl1  = (const float*)d_in[49]; const float* cl1 = (const float*)d_in[50];
  const float* Wh1  = (const float*)d_in[51]; const float* bh1 = (const float*)d_in[52];
  const float* gh1  = (const float*)d_in[53]; const float* ch1 = (const float*)d_in[54];
  const float* Wh2  = (const float*)d_in[55]; const float* bh2 = (const float*)d_in[56];
  const float* gh2  = (const float*)d_in[57]; const float* ch2 = (const float*)d_in[58];
  const float* Wout = (const float*)d_in[59]; const float* bout = (const float*)d_in[60];

  float* out = (float*)d_out; // [8,60] fp32 log-probs

  // ------------------------- workspace allocator --------------------------
  char* base = (char*)d_ws;
  size_t off = 0;
  auto ALLOC = [&](size_t bytes) -> char* {
    char* p = base + off;
    off += (bytes + 255) & ~(size_t)255;
    return p;
  };
  auto AH = [&](size_t elems) -> _Float16* { return (_Float16*)ALLOC(elems * 2); };
  auto AF = [&](size_t elems) -> float*    { return (float*)ALLOC(elems * 4); };

  // f16 weights (padded K where needed)
  _Float16* Wc1h  = AH(64 * 32);     _Float16* Wc2h  = AH(128 * 64);
  _Float16* Wc3h  = AH(1024 * 128);  _Float16* Wf1h  = AH(512 * 1024);
  _Float16* Wf2h  = AH(256 * 512);   _Float16* Wf3h  = AH(9 * 256);
  _Float16* Wm10h = AH(64 * 32);     _Float16* Wm11h = AH(64 * 64);
  _Float16* Wm12h = AH(64 * 64);     _Float16* Wm2h  = AH(128 * 128);
  _Float16* Wm3h  = AH(256 * 256);   _Float16* Wl1h  = AH(1024 * 448);
  _Float16* Wh1h  = AH(512 * 1024);  _Float16* Wh2h  = AH(256 * 512);
  _Float16* Wouth = AH(60 * 256);

  // activations / scratch
  _Float16* posh  = AH((size_t)NP * 32);
  _Float16* h1h   = AH((size_t)NP * 64);
  _Float16* h2h   = AH((size_t)NP * 128);
  float*    BIG   = AF((size_t)NP * 1024);    // h3, later feat
  float*    gvec  = AF(BB * 1024);
  _Float16* gvech = AH(BB * 1024);
  _Float16* s1h   = AH(BB * 512);
  _Float16* s2h   = AH(BB * 256);
  float*    t9    = AF(BB * 16);
  float*    posT  = AF((size_t)NP * 3);
  float*    x1    = AF((size_t)NP * 64);
  float*    x2    = AF((size_t)NP * 128);
  float*    x3    = AF((size_t)NP * 256);
  _Float16* fpad  = AH((size_t)NP * 160);
  float*    fsq   = AF(NP);
  int*      knn   = (int*)ALLOC((size_t)NP * KNB * 4);
  _Float16* Eh    = AH((size_t)NN * KNB * 256);
  float*    ebuf  = AF((size_t)NN * KNB * 256);
  _Float16* mida  = AH((size_t)NN * KNB * 64);
  _Float16* midb  = AH((size_t)NN * KNB * 64);
  _Float16* cath  = AH((size_t)NP * 448);
  float*    gfeat = AF(BB * 1024);
  _Float16* gfh   = AH(BB * 1024);
  _Float16* hh1   = AH(BB * 512);
  _Float16* hh2   = AH(BB * 256);
  float*    logit = AF(BB * 64);

  // ------------------------- weight conversions ---------------------------
  launch_cvt(stream, Wc1, Wc1h, 64, 3, 32);
  launch_cvt(stream, Wc2, Wc2h, 128, 64, 64);
  launch_cvt(stream, Wc3, Wc3h, 1024, 128, 128);
  launch_cvt(stream, Wf1, Wf1h, 512, 1024, 1024);
  launch_cvt(stream, Wf2, Wf2h, 256, 512, 512);
  launch_cvt(stream, Wf3, Wf3h, 9, 256, 256);
  launch_cvt(stream, Wm10, Wm10h, 64, 6, 32);
  launch_cvt(stream, Wm11, Wm11h, 64, 64, 64);
  launch_cvt(stream, Wm12, Wm12h, 64, 64, 64);
  launch_cvt(stream, Wm2, Wm2h, 128, 128, 128);
  launch_cvt(stream, Wm3, Wm3h, 256, 256, 256);
  launch_cvt(stream, Wl1, Wl1h, 1024, 448, 448);
  launch_cvt(stream, Wh1, Wh1h, 512, 1024, 1024);
  launch_cvt(stream, Wh2, Wh2h, 256, 512, 512);
  launch_cvt(stream, Wout, Wouth, 60, 256, 256);

  // ------------------------------- STN ------------------------------------
  launch_cvt(stream, pos, posh, NP, 3, 32);
  launch_gemm(stream, posh, Wc1h, bc1, g1, bb1, nullptr, h1h, NP, 64, 32, 2);
  launch_gemm(stream, h1h,  Wc2h, bc2, g2, bb2, nullptr, h2h, NP, 128, 64, 2);
  launch_gemm(stream, h2h,  Wc3h, bc3, g3, bb3, BIG, nullptr, NP, 1024, 128, 2);
  k_maxpool<<<(BB * 1024 + 255) / 256, 256, 0, stream>>>(BIG, gvec, 1024);
  launch_cvt(stream, gvec, gvech, BB, 1024, 1024);
  launch_gemm(stream, gvech, Wf1h, bf1, g4, bb4, nullptr, s1h, BB, 512, 1024, 2);
  launch_gemm(stream, s1h,   Wf2h, bf2, g5, bb5, nullptr, s2h, BB, 256, 512, 2);
  launch_gemm(stream, s2h,   Wf3h, bf3, nullptr, nullptr, t9, nullptr, BB, 9, 256, 0);
  k_addeye<<<1, 128, 0, stream>>>(t9);
  k_bmm<<<(NP + 255) / 256, 256, 0, stream>>>(pos, t9, posT);

  // ---------------------------- edge conv 1 -------------------------------
  k_feat<<<(NP + 255) / 256, 256, 0, stream>>>(posT, 3, seq, fpad, 32, fsq);
  k_knn<32><<<(NP + 127) / 128, 128, 0, stream>>>(fpad, fsq, knn);
  for (int ch = 0; ch < BB; ++ch) {
    int pt0 = ch * NN;
    k_gather<<<(NN * KNB + 255) / 256, 256, 0, stream>>>(posT, 3, knn, pt0, Eh, 32);
    launch_gemm(stream, Eh,   Wm10h, bm10, gm10, cm10, nullptr, mida, NN * KNB, 64, 32, 1);
    launch_gemm(stream, mida, Wm11h, bm11, gm11, cm11, nullptr, midb, NN * KNB, 64, 64, 1);
    launch_gemm(stream, midb, Wm12h, bm12, gm12, cm12, ebuf, nullptr, NN * KNB, 64, 64, 1);
    k_att<64><<<NN / 8, 256, 0, stream>>>(ebuf, att1, x1, pt0);
  }

  // ---------------------------- edge conv 2 -------------------------------
  k_feat<<<(NP + 255) / 256, 256, 0, stream>>>(x1, 64, seq, fpad, 96, fsq);
  k_knn<96><<<(NP + 127) / 128, 128, 0, stream>>>(fpad, fsq, knn);
  for (int ch = 0; ch < BB; ++ch) {
    int pt0 = ch * NN;
    k_gather<<<(NN * KNB + 255) / 256, 256, 0, stream>>>(x1, 64, knn, pt0, Eh, 128);
    launch_gemm(stream, Eh, Wm2h, bm2, gm2, cm2, ebuf, nullptr, NN * KNB, 128, 128, 1);
    k_att<128><<<NN / 8, 256, 0, stream>>>(ebuf, att2, x2, pt0);
  }

  // ---------------------------- edge conv 3 -------------------------------
  k_feat<<<(NP + 255) / 256, 256, 0, stream>>>(x2, 128, seq, fpad, 160, fsq);
  k_knn<160><<<(NP + 127) / 128, 128, 0, stream>>>(fpad, fsq, knn);
  for (int ch = 0; ch < BB; ++ch) {
    int pt0 = ch * NN;
    k_gather<<<(NN * KNB + 255) / 256, 256, 0, stream>>>(x2, 128, knn, pt0, Eh, 256);
    launch_gemm(stream, Eh, Wm3h, bm3, gm3, cm3, ebuf, nullptr, NN * KNB, 256, 256, 1);
    k_att<256><<<NN / 8, 256, 0, stream>>>(ebuf, att3, x3, pt0);
  }

  // ------------------------- lin1 + pool + head ---------------------------
  k_concat<<<((size_t)NP * 448 + 255) / 256, 256, 0, stream>>>(x1, x2, x3, cath);
  launch_gemm(stream, cath, Wl1h, bl1, gl1, cl1, BIG, nullptr, NP, 1024, 448, 1);
  k_maxpool<<<(BB * 1024 + 255) / 256, 256, 0, stream>>>(BIG, gfeat, 1024);
  launch_cvt(stream, gfeat, gfh, BB, 1024, 1024);
  launch_gemm(stream, gfh, Wh1h, bh1, gh1, ch1, nullptr, hh1, BB, 512, 1024, 1);
  launch_gemm(stream, hh1, Wh2h, bh2, gh2, ch2, nullptr, hh2, BB, 256, 512, 1);
  launch_gemm(stream, hh2, Wouth, bout, nullptr, nullptr, logit, nullptr, BB, 60, 256, 0);
  k_logsoftmax<<<1, BB, 0, stream>>>(logit, out);
}
